// Encoder_81819126989050
// MI455X (gfx1250) — matrix operand order repaired and checked by reading, not compiled, not run on hardware
//
#include <hip/hip_runtime.h>

// ---------------------------------------------------------------------------
// Problem constants (from reference)
// ---------------------------------------------------------------------------
#define NN   20000      // nodes
#define NE   320000     // edges
#define FIN  256
#define FHID 256
#define FOUT 128
#define FEO  64

typedef __attribute__((ext_vector_type(16))) __bf16 v16bf;
typedef __attribute__((ext_vector_type(8)))  __bf16 v8bf;
typedef __attribute__((ext_vector_type(8)))  float  v8f;

// Assemble a 16-element bf16 fragment from two 16B-aligned LDS halves.
__device__ __forceinline__ v16bf load_frag16(const __bf16* p) {
    v8bf lo = *(const v8bf*)(p);
    v8bf hi = *(const v8bf*)(p + 8);
    v16bf r;
#pragma unroll
    for (int i = 0; i < 8; ++i) { r[i] = lo[i]; r[8 + i] = hi[i]; }
    return r;
}

// ---------------------------------------------------------------------------
// WMMA GEMM: C[M,Nc] = A[M,K] @ B[K,Nc] (+ optional bias[Nc]) ; row-major f32,
// bf16 compute. Block = 256 threads = 8 wave32, tile 128(M) x 32(N).
// Wave w owns 16-row strip (blockIdx.y*8 + w) and both 16-wide N subtiles.
// B tile (32x32) is staged to LDS per k-step, converted to bf16 in WMMA
// fragment order: [s in 0..1][lane 0..31] -> 16 bf16 at stride 24 elem (48B).
// Fragment layouts per CDNA5 ISA 7.12.2 (wave32):
//   A 16x32 bf16 : lane l -> row m=l&15; K runs [8h..8h+7], [16+8h..16+8h+7]
//   B 32x16 bf16 : lane l -> col n=l&15; K runs [8h..8h+7], [16+8h..16+8h+7]
//   C/D 16x16 f32: VGPR r -> row r + 8h, col n=l&15          (h = l>>4)
// ---------------------------------------------------------------------------
__global__ __launch_bounds__(256)
void gemm_wmma_kernel(const float* __restrict__ A, const float* __restrict__ B,
                      const float* __restrict__ bias, float* __restrict__ C,
                      int M, int K, int lda, int ldb, int ldc) {
    __shared__ __align__(64) __bf16 bstage[2 * 32 * 24];   // 3 KB

    const int tid   = threadIdx.x;
    const int lane  = tid & 31;
    const int wave  = tid >> 5;
    const int n0    = blockIdx.x << 5;             // 32-wide N tile
    const int strip = blockIdx.y * 8 + wave;       // 16-row M strip
    const bool valid = (strip << 4) < M;
    const int m0    = strip << 4;

    const int mr   = lane & 15;    // A row in tile == B/C col in subtile
    const int half = lane >> 4;

    // staging coords: one float4 of the 32x32 B tile per thread per k-step
    const int sr  = tid >> 3;            // k within tile, 0..31
    const int sq  = (tid & 7) << 2;      // n within tile, 0,4,...,28
    const int se  = (sr & 7) | ((sr >> 4) << 3);   // fragment element index
    const int skh = ((sr >> 3) & 1) << 4;          // k bit 3 -> lane bit 4

    v8f acc0 = {}, acc1 = {};
    const float* Arow = A + (size_t)(valid ? (m0 + mr) : 0) * lda;

    for (int k0 = 0; k0 < K; k0 += 32) {
        // ---- cooperative B staging: global float4 -> bf16 fragment order ----
        const float4 bq = *(const float4*)(B + (size_t)(k0 + sr) * ldb + n0 + sq);
        {
            float v[4] = {bq.x, bq.y, bq.z, bq.w};
#pragma unroll
            for (int i = 0; i < 4; ++i) {
                int nl = sq + i;                        // 0..31 within tile
                int s  = nl >> 4;                       // subtile
                int bl = (nl & 15) | skh;               // fragment lane
                bstage[(s * 32 + bl) * 24 + se] = (__bf16)v[i];
            }
        }
        __syncthreads();

        if (valid) {
            // ---- A fragment: two contiguous 8-float runs -> 16 bf16 ----
            const float* ap = Arow + k0 + (half << 3);
            float4 a0 = *(const float4*)(ap + 0);
            float4 a1 = *(const float4*)(ap + 4);
            float4 a2 = *(const float4*)(ap + 16);
            float4 a3 = *(const float4*)(ap + 20);
            if (k0 + 32 < K) __builtin_prefetch(ap + 32, 0, 3);  // global_prefetch
            v16bf af;
            af[0]  = (__bf16)a0.x; af[1]  = (__bf16)a0.y; af[2]  = (__bf16)a0.z; af[3]  = (__bf16)a0.w;
            af[4]  = (__bf16)a1.x; af[5]  = (__bf16)a1.y; af[6]  = (__bf16)a1.z; af[7]  = (__bf16)a1.w;
            af[8]  = (__bf16)a2.x; af[9]  = (__bf16)a2.y; af[10] = (__bf16)a2.z; af[11] = (__bf16)a2.w;
            af[12] = (__bf16)a3.x; af[13] = (__bf16)a3.y; af[14] = (__bf16)a3.z; af[15] = (__bf16)a3.w;

            v16bf b0 = load_frag16(&bstage[(0 * 32 + lane) * 24]);
            v16bf b1 = load_frag16(&bstage[(1 * 32 + lane) * 24]);

            acc0 = __builtin_amdgcn_wmma_f32_16x16x32_bf16(
                       false, af, false, b0, (short)0, acc0, false, false);
            acc1 = __builtin_amdgcn_wmma_f32_16x16x32_bf16(
                       false, af, false, b1, (short)0, acc1, false, false);
        }
        __syncthreads();
    }

    if (valid) {
        const float bv0 = bias ? bias[n0 + mr]      : 0.0f;
        const float bv1 = bias ? bias[n0 + 16 + mr] : 0.0f;
        float* crow = C + (size_t)(m0 + (half << 3)) * ldc + n0 + mr;
#pragma unroll
        for (int r = 0; r < 8; ++r) {
            crow[(size_t)r * ldc]      = acc0[r] + bv0;
            crow[(size_t)r * ldc + 16] = acc1[r] + bv1;
        }
    }
}

// ---------------------------------------------------------------------------
// Degree / normalization
// ---------------------------------------------------------------------------
__global__ void init_deg_kernel(float* deg, int n) {
    int i = blockIdx.x * blockDim.x + threadIdx.x;
    if (i < n) deg[i] = 1.0f + 1e-6f;   // self-loop contribution + eps
}
__global__ void deg_edge_kernel(const int* __restrict__ col, float* deg, int e) {
    int i = blockIdx.x * blockDim.x + threadIdx.x;
    if (i < e) atomicAdd(&deg[col[i]], 1.0f);
}
__global__ void finalize_dis_kernel(float* deg, int n) {
    int i = blockIdx.x * blockDim.x + threadIdx.x;
    if (i < n) deg[i] = rsqrtf(deg[i]);
}

// v[f] = sum_k ew[k] * new_w[F + k, f]   (edge-scalar projection vector)
__global__ void make_vvec_kernel(const float* __restrict__ ew,
                                 const float* __restrict__ neww,
                                 float* __restrict__ v, int F) {
    int f = blockIdx.x * blockDim.x + threadIdx.x;
    if (f >= F) return;
    float s = 0.0f;
#pragma unroll 8
    for (int k = 0; k < FEO; ++k) s += ew[k] * neww[(size_t)(F + k) * F + f];
    v[f] = s;
}

__global__ void zero_kernel(float4* p, long long n4) {
    long long i = (long long)blockIdx.x * blockDim.x + threadIdx.x;
    if (i < n4) p[i] = make_float4(0.f, 0.f, 0.f, 0.f);
}

// ---------------------------------------------------------------------------
// Edge scatter: agg[col[e]] += dis[row]*dis[col] * (t[row[e]] + ea[e]*vvec)
// thread = (edge, 4-feature chunk); destination is L2-resident -> f32 atomics
// ---------------------------------------------------------------------------
__global__ void scatter_kernel(const int* __restrict__ rowi, const int* __restrict__ coli,
                               const float* __restrict__ ea, const float* __restrict__ dis,
                               const float* __restrict__ t, const float* __restrict__ vvec,
                               float* __restrict__ agg, int E, int F, int lgc) {
    long long idx = (long long)blockIdx.x * blockDim.x + threadIdx.x;
    long long total = (long long)E << lgc;
    if (idx >= total) return;
    int e = (int)(idx >> lgc);
    int c = ((int)idx & ((1 << lgc) - 1)) << 2;
    int j = rowi[e], i = coli[e];
    float w = dis[j] * dis[i];
    float a = ea[e];
    float4 tv = *(const float4*)(t + (size_t)j * F + c);
    float4 vv = *(const float4*)(vvec + c);
    float* dst = agg + (size_t)i * F + c;
    atomicAdd(dst + 0, w * (tv.x + a * vv.x));
    atomicAdd(dst + 1, w * (tv.y + a * vv.y));
    atomicAdd(dst + 2, w * (tv.z + a * vv.z));
    atomicAdd(dst + 3, w * (tv.w + a * vv.w));
}

// Self loops: agg[i] += dis[i]^2 * t[i]   (exclusive per element, no atomics)
__global__ void selfloop_kernel(const float* __restrict__ dis, const float* __restrict__ t,
                                float* __restrict__ agg, int n, int F, int lgc) {
    long long idx = (long long)blockIdx.x * blockDim.x + threadIdx.x;
    long long total = (long long)n << lgc;
    if (idx >= total) return;
    int i = (int)(idx >> lgc);
    int c = ((int)idx & ((1 << lgc) - 1)) << 2;
    float w = dis[i] * dis[i];
    float4 tv = *(const float4*)(t + (size_t)i * F + c);
    float4* dst = (float4*)(agg + (size_t)i * F + c);
    float4 d = *dst;
    d.x += w * tv.x; d.y += w * tv.y; d.z += w * tv.z; d.w += w * tv.w;
    *dst = d;
}

// ---------------------------------------------------------------------------
// BatchNorm (training-mode batch stats): column mean/var then fused apply
// ---------------------------------------------------------------------------
__global__ void stats_kernel(const float* __restrict__ x, float* __restrict__ mean,
                             float* __restrict__ var, int n, int F) {
    int f = blockIdx.x;
    int tid = threadIdx.x;
    float s = 0.f, s2 = 0.f;
    for (int r = tid; r < n; r += blockDim.x) {
        float v = x[(size_t)r * F + f];
        s += v; s2 += v * v;
    }
    __shared__ float sh[512];
    sh[tid] = s; sh[256 + tid] = s2;
    __syncthreads();
    for (int off = 128; off > 0; off >>= 1) {
        if (tid < off) { sh[tid] += sh[tid + off]; sh[256 + tid] += sh[256 + tid + off]; }
        __syncthreads();
    }
    if (tid == 0) {
        float m = sh[0] / n;
        mean[f] = m;
        var[f] = sh[256] / n - m * m;   // ddof=0 to match jnp.var
    }
}

// out = [relu]( (x - m) * rsqrt(v+1e-5) * g + b ) [+ res]
__global__ void bn_apply_kernel(const float* __restrict__ x, const float* __restrict__ mean,
                                const float* __restrict__ var, const float* __restrict__ g,
                                const float* __restrict__ b, const float* __restrict__ res,
                                float* __restrict__ out, long long total, int Fmask, int relu) {
    long long idx = (long long)blockIdx.x * blockDim.x + threadIdx.x;
    if (idx >= total) return;
    int f = (int)idx & Fmask;
    float v = (x[idx] - mean[f]) * rsqrtf(var[f] + 1e-5f) * g[f] + b[f];
    if (relu) v = fmaxf(v, 0.0f);
    if (res)  v += res[idx];
    out[idx] = v;
}

// ---------------------------------------------------------------------------
// Host-side launch
// ---------------------------------------------------------------------------
static inline void launch_gemm(const float* A, const float* B, const float* bias,
                               float* C, int M, int K, int Nc, hipStream_t s) {
    dim3 g(Nc / 32, (M / 16 + 7) / 8);
    gemm_wmma_kernel<<<g, 256, 0, s>>>(A, B, bias, C, M, K, K, Nc, Nc);
}
static inline void launch_zero(float* p, long long n, hipStream_t s) {
    long long n4 = n >> 2;
    zero_kernel<<<(unsigned)((n4 + 255) / 256), 256, 0, s>>>((float4*)p, n4);
}

extern "C" void kernel_launch(void* const* d_in, const int* in_sizes, int n_in,
                              void* d_out, int out_size, void* d_ws, size_t ws_size,
                              hipStream_t stream) {
    (void)in_sizes; (void)n_in; (void)out_size; (void)ws_size;

    const float* x      = (const float*)d_in[0];
    const int*   ei     = (const int*)  d_in[1];
    const float* eattr  = (const float*)d_in[2];
    const float* c1_nw  = (const float*)d_in[3];
    const float* c1_ew  = (const float*)d_in[4];
    const float* c1_new = (const float*)d_in[5];
    const float* c2_nw  = (const float*)d_in[7];
    const float* c2_ew  = (const float*)d_in[8];
    const float* c2_new = (const float*)d_in[9];
    const float* cm_nw  = (const float*)d_in[11];
    const float* cm_ew  = (const float*)d_in[12];
    const float* cm_new = (const float*)d_in[13];
    const float* cl_nw  = (const float*)d_in[15];
    const float* cl_ew  = (const float*)d_in[16];
    const float* cl_new = (const float*)d_in[17];
    const float* lin1_w = (const float*)d_in[19];
    const float* lin1_b = (const float*)d_in[20];
    const float* lin2_w = (const float*)d_in[21];
    const float* lin2_b = (const float*)d_in[22];
    const float* pih_w  = (const float*)d_in[23];
    const float* pih_b  = (const float*)d_in[24];
    const float* pho_w  = (const float*)d_in[25];
    const float* pho_b  = (const float*)d_in[26];
    const float* bn1_g  = (const float*)d_in[27];
    const float* bn1_b  = (const float*)d_in[28];
    const float* bn2_g  = (const float*)d_in[29];
    const float* bn2_b  = (const float*)d_in[30];
    const float* bnm_g  = (const float*)d_in[31];
    const float* bnm_b  = (const float*)d_in[32];
    const float* bnl_g  = (const float*)d_in[33];
    const float* bnl_b  = (const float*)d_in[34];
    // conv biases (d_in[6,10,14,18]) are absorbed by the following BatchNorm.

    const int* row = ei;        // edge_index[0], source j
    const int* col = ei + NE;   // edge_index[1], target i

    float* out_mean   = (float*)d_out;
    float* out_logstd = out_mean + (size_t)NN * FOUT;

    // ---- workspace layout ----
    float* W = (float*)d_ws;
    size_t o = 0;
    float* dis  = W + o; o += NN;
    float* Wc1  = W + o; o += 256 * 256;
    float* Wc2  = W + o; o += 256 * 256;
    float* Wcm  = W + o; o += 128 * 128;
    float* Wcl  = W + o; o += 128 * 128;
    float* v1   = W + o; o += 256;
    float* v2   = W + o; o += 256;
    float* vm   = W + o; o += 128;
    float* vl   = W + o; o += 128;
    float* mbuf = W + o; o += 256;
    float* vbuf = W + o; o += 256;
    o = (o + 255) & ~(size_t)255;
    float* bufA = W + o; o += (size_t)NN * 256;
    float* bufB = W + o; o += (size_t)NN * 256;
    float* bufC = W + o; o += (size_t)NN * 256;
    float* bufD = W + o; o += (size_t)NN * 256;

    const long long NF256 = (long long)NN * 256;
    const long long NF128 = (long long)NN * 128;

    // ---- degrees -> dis = deg^-0.5 ----
    init_deg_kernel<<<(NN + 255) / 256, 256, 0, stream>>>(dis, NN);
    deg_edge_kernel<<<(NE + 255) / 256, 256, 0, stream>>>(col, dis, NE);
    finalize_dis_kernel<<<(NN + 255) / 256, 256, 0, stream>>>(dis, NN);

    // ---- fold conv weights: Wc = nw @ new_w_top ; v = ew @ new_w_bot ----
    launch_gemm(c1_nw, c1_new, nullptr, Wc1, 256, 256, 256, stream);
    launch_gemm(c2_nw, c2_new, nullptr, Wc2, 256, 256, 256, stream);
    launch_gemm(cm_nw, cm_new, nullptr, Wcm, 128, 128, 128, stream);
    launch_gemm(cl_nw, cl_new, nullptr, Wcl, 128, 128, 128, stream);
    make_vvec_kernel<<<1, 256, 0, stream>>>(c1_ew, c1_new, v1, 256);
    make_vvec_kernel<<<1, 256, 0, stream>>>(c2_ew, c2_new, v2, 256);
    make_vvec_kernel<<<1, 128, 0, stream>>>(cm_ew, cm_new, vm, 128);
    make_vvec_kernel<<<1, 128, 0, stream>>>(cl_ew, cl_new, vl, 128);

    // ================= conv1 + bn + relu + residual(x) -> x1 (bufC) ========
    launch_gemm(x, Wc1, nullptr, bufA, NN, 256, 256, stream);          // t1
    launch_zero(bufB, NF256, stream);
    scatter_kernel<<<(unsigned)(((long long)NE * 64 + 255) / 256), 256, 0, stream>>>(
        row, col, eattr, dis, bufA, v1, bufB, NE, 256, 6);
    selfloop_kernel<<<(unsigned)(((long long)NN * 64 + 255) / 256), 256, 0, stream>>>(
        dis, bufA, bufB, NN, 256, 6);
    stats_kernel<<<256, 256, 0, stream>>>(bufB, mbuf, vbuf, NN, 256);
    bn_apply_kernel<<<(unsigned)((NF256 + 255) / 256), 256, 0, stream>>>(
        bufB, mbuf, vbuf, bn1_g, bn1_b, x, bufC, NF256, 255, 1);

    // ================= lin1 -> conv2 input (bufA) ==========================
    launch_gemm(bufC, lin1_w, lin1_b, bufA, NN, 256, 256, stream);
    // conv2
    launch_gemm(bufA, Wc2, nullptr, bufB, NN, 256, 256, stream);       // t2
    launch_zero(bufD, NF256, stream);
    scatter_kernel<<<(unsigned)(((long long)NE * 64 + 255) / 256), 256, 0, stream>>>(
        row, col, eattr, dis, bufB, v2, bufD, NE, 256, 6);
    selfloop_kernel<<<(unsigned)(((long long)NN * 64 + 255) / 256), 256, 0, stream>>>(
        dis, bufB, bufD, NN, 256, 6);
    launch_gemm(bufC, pih_w, pih_b, bufA, NN, 256, 256, stream);       // pih residual
    stats_kernel<<<256, 256, 0, stream>>>(bufD, mbuf, vbuf, NN, 256);
    bn_apply_kernel<<<(unsigned)((NF256 + 255) / 256), 256, 0, stream>>>(
        bufD, mbuf, vbuf, bn2_g, bn2_b, bufA, bufB, NF256, 255, 1);    // x2 -> bufB

    // ================= lin2 -> x3 (bufC, N x 128) ==========================
    launch_gemm(bufB, lin2_w, lin2_b, bufC, NN, 256, 128, stream);

    // ================= conv_mean path -> d_out[0 : N*128) ==================
    launch_gemm(bufC, Wcm, nullptr, bufD, NN, 128, 128, stream);       // tm
    launch_zero(bufA, NF128, stream);
    scatter_kernel<<<(unsigned)(((long long)NE * 32 + 255) / 256), 256, 0, stream>>>(
        row, col, eattr, dis, bufD, vm, bufA, NE, 128, 5);
    selfloop_kernel<<<(unsigned)(((long long)NN * 32 + 255) / 256), 256, 0, stream>>>(
        dis, bufD, bufA, NN, 128, 5);
    float* pho = bufD + NF128;
    launch_gemm(bufB, pho_w, pho_b, pho, NN, 256, 128, stream);        // pho residual
    stats_kernel<<<128, 256, 0, stream>>>(bufA, mbuf, vbuf, NN, 128);
    bn_apply_kernel<<<(unsigned)((NF128 + 255) / 256), 256, 0, stream>>>(
        bufA, mbuf, vbuf, bnm_g, bnm_b, pho, out_mean, NF128, 127, 0);

    // ================= conv_logstd path -> d_out[N*128 : ) =================
    launch_gemm(bufC, Wcl, nullptr, bufD, NN, 128, 128, stream);       // tl
    launch_zero(bufA, NF128, stream);
    scatter_kernel<<<(unsigned)(((long long)NE * 32 + 255) / 256), 256, 0, stream>>>(
        row, col, eattr, dis, bufD, vl, bufA, NE, 128, 5);
    selfloop_kernel<<<(unsigned)(((long long)NN * 32 + 255) / 256), 256, 0, stream>>>(
        dis, bufD, bufA, NN, 128, 5);
    stats_kernel<<<128, 256, 0, stream>>>(bufA, mbuf, vbuf, NN, 128);
    bn_apply_kernel<<<(unsigned)((NF128 + 255) / 256), 256, 0, stream>>>(
        bufA, mbuf, vbuf, bnl_g, bnl_b, nullptr, bufD, NF128, 127, 0); // L -> bufD
    // faithful double-BN: logstd = L + bn(L)
    stats_kernel<<<128, 256, 0, stream>>>(bufD, mbuf, vbuf, NN, 128);
    bn_apply_kernel<<<(unsigned)((NF128 + 255) / 256), 256, 0, stream>>>(
        bufD, mbuf, vbuf, bnl_g, bnl_b, bufD, out_logstd, NF128, 127, 0);
}
